// VectorQuantizer_13443247636961
// MI455X (gfx1250) — compile-verified, matrix operand order checked by reading, not run on hardware
//
#include <hip/hip_runtime.h>

typedef _Float16 v16h __attribute__((ext_vector_type(16)));
typedef float    v8f  __attribute__((ext_vector_type(8)));
typedef int      v4i_b128 __attribute__((vector_size(16)));   // matches builtin param pointee

#define NCODES 1024
#define CDIM   256
#define HW     4096                 // 64*64
#define NVEC   131072               // 32*64*64
#define QELEMS 33554432             // 32*256*64*64
#define ROWS   64                   // rows per block (4 waves * 16)
#define LSTR   264                  // LDS row stride in f16 (256 + 8 pad)
#define TILE_F16 4096               // one B tile: 16 codes x 256 k = 8KB f16

// ---------------- workspace layout ----------------
#define WS_FRAGB  0                        // 512KB: emb f16, pre-swizzled B fragments
#define WS_HNORM  (NCODES*CDIM*2)          // 4KB:   0.5*||e||^2
#define WS_COUNTS (WS_HNORM + NCODES*4)    // 4KB:   histogram
#define WS_LOSS   (WS_COUNTS + NCODES*4)   // 4B:    loss accumulator

#if defined(__has_builtin)
#if __has_builtin(__builtin_amdgcn_global_load_async_to_lds_b128) && \
    __has_builtin(__builtin_amdgcn_s_wait_asynccnt)
#define VQ_ASYNC 1
#endif
#endif
#ifndef VQ_ASYNC
#define VQ_ASYNC 0
#endif

#if VQ_ASYNC
typedef __attribute__((address_space(1))) v4i_b128* gas_b128_t;
typedef __attribute__((address_space(3))) v4i_b128* las_b128_t;
#endif

// (lane,e) -> k map within a 32-wide K tile (16-bit A/B fragment layout,
// CDNA5 ISA 7.12.2): lanes 0-15: e0-7 -> k0-7, e8-15 -> k16-23;
// lanes 16-31: e0-7 -> k8-15, e8-15 -> k24-31.

__global__ void vq_init(float* __restrict__ counts, float* __restrict__ loss) {
    int t = blockIdx.x * blockDim.x + threadIdx.x;
    if (t < NCODES) counts[t] = 0.0f;
    if (t == 0) *loss = 0.0f;
}

// One block per code; thread k handles channel k.
__global__ __launch_bounds__(256)
void vq_prep(const float* __restrict__ emb, _Float16* __restrict__ fragB,
             float* __restrict__ hnorm) {
    int n = blockIdx.x;
    int k = threadIdx.x;
    float v = emb[n * CDIM + k];
    int ntile = n >> 4, nl = n & 15;
    int kstep = k >> 5, k32 = k & 31;
    int lane = nl + (((k32 >> 3) & 1) << 4);
    int e    = (k32 & 7) + (((k32 >> 4) & 1) << 3);
    fragB[((((ntile << 3) + kstep) * 32 + lane) << 4) + e] = (_Float16)v;

    __shared__ float red[256];
    red[k] = v * v;
    __syncthreads();
    for (int s = 128; s > 0; s >>= 1) {
        if (k < s) red[k] += red[k + s];
        __syncthreads();
    }
    if (k == 0) hnorm[n] = 0.5f * red[0];
}

__global__ __launch_bounds__(128)
void vq_main(const float* __restrict__ z, const float* __restrict__ emb,
             const _Float16* __restrict__ fragB, const float* __restrict__ hnorm,
             float* __restrict__ out_quant, float* __restrict__ out_codes,
             float* __restrict__ counts, float* __restrict__ loss_acc) {
    __shared__ __align__(16) _Float16 ldsA[ROWS * LSTR];   // 33792 B
    __shared__ __align__(16) _Float16 ldsB[2][TILE_F16];   // 16384 B (double-buffered B tile)
    __shared__ float snorm[NCODES];                        //  4096 B
    __shared__ float red_val[4][16][16];                   //  4096 B
    __shared__ int   red_idx[4][16][16];                   //  4096 B
    __shared__ int   scode[ROWS];                          //   256 B
    __shared__ float lsum[128];                            //   512 B

    const int tid = threadIdx.x;
    const int n0  = blockIdx.x * ROWS;
    const int b   = n0 >> 12;                   // 4096 pixels per image
    const int hw0 = n0 & (HW - 1);              // 64 | 4096 -> single image per block
    const float* zb = z + (size_t)b * CDIM * HW + hw0;

#if VQ_ASYNC
    // async copy of one 8KB B tile: 128 threads x 4 x b128 = 8KB, ASYNCcnt += 4/wave
#define ISSUE_TILE(nt_, buf_)                                                        \
    {                                                                                \
        uintptr_t _g = (uintptr_t)(fragB + (size_t)(nt_) * TILE_F16);                \
        uintptr_t _l = (uintptr_t)(&ldsB[(buf_)][0]);                                \
        _Pragma("unroll")                                                            \
        for (int _j = 0; _j < 4; ++_j) {                                             \
            __builtin_amdgcn_global_load_async_to_lds_b128(                          \
                (gas_b128_t)(_g + tid * 16 + _j * 2048),                             \
                (las_b128_t)(_l + tid * 16 + _j * 2048),                             \
                0, 0);                                                               \
        }                                                                            \
    }
#endif

    // stage 0.5*||e||^2
    for (int i = tid; i < NCODES; i += 128) snorm[i] = hnorm[i];

    // stage the 64x256 z tile as f16 into LDS; float4 over the contiguous row dim
    for (int i = tid; i < (ROWS * CDIM) / 4; i += 128) {
        int c = i >> 4, rg = i & 15;
        float4 v = *reinterpret_cast<const float4*>(zb + c * HW + rg * 4);
        int r = rg * 4;
        ldsA[(r + 0) * LSTR + c] = (_Float16)v.x;
        ldsA[(r + 1) * LSTR + c] = (_Float16)v.y;
        ldsA[(r + 2) * LSTR + c] = (_Float16)v.z;
        ldsA[(r + 3) * LSTR + c] = (_Float16)v.w;
    }

#if VQ_ASYNC
    ISSUE_TILE(0, 0);                   // prefetch first two B tiles
    ISSUE_TILE(1, 1);
#endif
    __syncthreads();                    // A tile (and norms) visible

    const int wave = tid >> 5;
    const int lane = tid & 31;
    const int lgrp = lane >> 4;
    const int l16  = lane & 15;

    union Fr { v16h v; uint4 u[2]; };

    // preload this wave's 8 A k-fragments (16x32 f16 each) from LDS
    Fr a[8];
    const _Float16* arow = ldsA + (wave * 16 + l16) * LSTR;
    #pragma unroll
    for (int ks = 0; ks < 8; ++ks) {
        int kA = ks * 32 + (lgrp ? 8 : 0);
        int kB = ks * 32 + (lgrp ? 24 : 16);
        a[ks].u[0] = *reinterpret_cast<const uint4*>(arow + kA);
        a[ks].u[1] = *reinterpret_cast<const uint4*>(arow + kB);
    }

    float bv[8];
    int   bi[8];
    #pragma unroll
    for (int r = 0; r < 8; ++r) { bv[r] = -3.4e38f; bi[r] = 0; }

    for (int nt = 0; nt < 64; ++nt) {
#if VQ_ASYNC
        if (nt == 63) __builtin_amdgcn_s_wait_asynccnt(0);   // last tile: drain
        else          __builtin_amdgcn_s_wait_asynccnt(4);   // tile nt done, nt+1 in flight
        __syncthreads();                                     // all waves' portions visible
        const _Float16* bt = ldsB[nt & 1];
#else
        __syncthreads();                                     // prior readers done
        {
            const uint4* gsrc = reinterpret_cast<const uint4*>(fragB + (size_t)nt * TILE_F16);
            uint4*       ldst = reinterpret_cast<uint4*>(&ldsB[0][0]);
            #pragma unroll
            for (int j = 0; j < 4; ++j) ldst[tid + j * 128] = gsrc[tid + j * 128];
        }
        __syncthreads();
        const _Float16* bt = ldsB[0];
#endif
        v8f acc = {};
        const _Float16* bp = bt + lane * 16;
        #pragma unroll
        for (int ks = 0; ks < 8; ++ks) {
            Fr bf;
            bf.u[0] = *reinterpret_cast<const uint4*>(bp);
            bf.u[1] = *reinterpret_cast<const uint4*>(bp + 8);
            bp += 32 * 16;
            acc = __builtin_amdgcn_wmma_f32_16x16x32_f16(
                      false, a[ks].v, false, bf.v, (short)0, acc, false, false);
        }
        int code = (nt << 4) + l16;
        float hn = snorm[code];
        #pragma unroll
        for (int r = 0; r < 8; ++r) {
            float s = acc[r] - hn;              // argmax(x.e - 0.5||e||^2) == argmin dist
            if (s > bv[r]) { bv[r] = s; bi[r] = code; }
        }
#if VQ_ASYNC
        __syncthreads();                        // everyone done reading ldsB[nt&1]
        if (nt + 2 < 64) ISSUE_TILE(nt + 2, nt & 1);
#endif
    }

    // per-lane bests -> LDS, reduce 16 candidates per row
    #pragma unroll
    for (int r = 0; r < 8; ++r) {
        int m = lgrp ? (r + 8) : r;             // C/D layout: lanes 16-31 hold rows 8-15
        red_val[wave][m][l16] = bv[r];
        red_idx[wave][m][l16] = bi[r];
    }
    __syncthreads();

    if (tid < ROWS) {
        int w = tid >> 4, m = tid & 15;
        float best = red_val[w][m][0];
        int   bidx = red_idx[w][m][0];
        for (int l = 1; l < 16; ++l) {          // ascending code order: '>' keeps lowest index
            float v = red_val[w][m][l];
            if (v > best) { best = v; bidx = red_idx[w][m][l]; }
        }
        int rr = (w << 4) + m;
        scode[rr] = bidx;
        out_codes[n0 + rr] = (float)bidx;
        atomicAdd(&counts[bidx], 1.0f);
    }
    __syncthreads();

    // gather quant rows (f32 codebook, L2-resident), write coalesced, accumulate MSE
    float sum = 0.0f;
    {
        int i  = tid & (ROWS - 1);
        int cg = tid >> 6;
        int k  = scode[i];
        const float* erow = emb + k * CDIM;
        float* qb = out_quant + (size_t)b * CDIM * HW + hw0;
        for (int c = cg; c < CDIM; c += 2) {
            float q  = erow[c];
            float zv = zb[c * HW + i];
            float d  = q - zv;
            sum += d * d;
            qb[c * HW + i] = q;
        }
    }
    lsum[tid] = sum;
    __syncthreads();
    for (int s = 64; s > 0; s >>= 1) {
        if (tid < s) lsum[tid] += lsum[tid + s];
        __syncthreads();
    }
    if (tid == 0) atomicAdd(loss_acc, lsum[0]);
}

__global__ __launch_bounds__(256)
void vq_final(const float* __restrict__ counts, const float* __restrict__ loss_acc,
              float* __restrict__ out) {
    __shared__ float red[256];
    int t = threadIdx.x;
    float e = 0.0f;
    for (int k = t; k < NCODES; k += 256) {
        float p = counts[k] * (1.0f / (float)NVEC);
        e += p * logf(p + 1e-10f);
    }
    red[t] = e;
    __syncthreads();
    for (int s = 128; s > 0; s >>= 1) {
        if (t < s) red[t] += red[t + s];
        __syncthreads();
    }
    if (t == 0) {
        out[QELEMS]     = 1.25f * (*loss_acc) / (float)QELEMS;  // stop-grads cancel at value level
        out[QELEMS + 1] = expf(-red[0]);                        // perplexity
    }
}

extern "C" void kernel_launch(void* const* d_in, const int* in_sizes, int n_in,
                              void* d_out, int out_size, void* d_ws, size_t ws_size,
                              hipStream_t stream) {
    const float* z   = (const float*)d_in[0];   // [32,256,64,64] f32
    const float* emb = (const float*)d_in[1];   // [1024,256]     f32
    float* out = (float*)d_out;                 // quant | loss | perp | codes(as f32)

    char* ws = (char*)d_ws;
    _Float16* fragB  = (_Float16*)(ws + WS_FRAGB);
    float*    hnorm  = (float*)(ws + WS_HNORM);
    float*    counts = (float*)(ws + WS_COUNTS);
    float*    loss   = (float*)(ws + WS_LOSS);

    float* out_quant = out;
    float* out_codes = out + QELEMS + 2;

    vq_init<<<4, 256, 0, stream>>>(counts, loss);
    vq_prep<<<NCODES, 256, 0, stream>>>(emb, fragB, hnorm);
    vq_main<<<NVEC / ROWS, 128, 0, stream>>>(z, emb, fragB, hnorm,
                                             out_quant, out_codes, counts, loss);
    vq_final<<<1, 256, 0, stream>>>(counts, loss, out);
}
